// RecurrentConvBlock_84335977824587
// MI455X (gfx1250) — matrix operand order repaired and checked by reading, not compiled, not run on hardware
//
#include <hip/hip_runtime.h>
#include <hip/hip_bf16.h>

typedef __attribute__((ext_vector_type(16))) __bf16 v16bf;
typedef __attribute__((ext_vector_type(8)))  float  v8f;
typedef __attribute__((ext_vector_type(4)))  unsigned int v4u;
typedef __attribute__((ext_vector_type(8)))  unsigned int v8u;

__device__ __forceinline__ unsigned short f32_to_bf16(float f) {
  union { float f; unsigned int u; } v; v.f = f;
  unsigned int u = v.u;
  u += 0x7FFFu + ((u >> 16) & 1u);   // round-to-nearest-even
  return (unsigned short)(u >> 16);
}

__device__ __forceinline__ float hsig(float x) {
  return fminf(fmaxf(0.2f * x + 0.5f, 0.0f), 1.0f);
}

// -------------------------------------------------------------------------
// One-time conversion kernels
// -------------------------------------------------------------------------
__global__ __launch_bounds__(256) void cast_f32_to_bf16_kernel(
    const float* __restrict__ in, unsigned short* __restrict__ out, int n) {
  int i = blockIdx.x * 256 + threadIdx.x;
  if (i < n) out[i] = f32_to_bf16(in[i]);
}

// fp32 [KLEN][256] -> bf16 [256][KLEN] (n-major, for contiguous B staging)
__global__ __launch_bounds__(256) void transpose_w_to_bf16_kernel(
    const float* __restrict__ in, unsigned short* __restrict__ out, int klen) {
  int i = blockIdx.x * 256 + threadIdx.x;
  if (i < klen * 256) {
    int k = i >> 8, n = i & 255;
    out[n * klen + k] = f32_to_bf16(in[i]);
  }
}

__global__ __launch_bounds__(256) void fill_zero_kernel(
    unsigned int* __restrict__ p, int nwords) {
  int i = blockIdx.x * 256 + threadIdx.x;
  if (i < nwords) p[i] = 0u;
}

// -------------------------------------------------------------------------
// Implicit-GEMM conv (SAME, stride 1) via bf16 WMMA, fp32 accumulate.
//
// Block = 512 threads (16 waves).  Output tile: 64 pixels (one image row,
// W==64) x all 256 channels.  Waves arranged 4 (M) x 4 (N); each wave owns
// a 16x64 slice = 4 f32 16x16 accumulators.
//
// Input staging: the KH input rows this block needs are bulk-DMA'd into an
// LDS row cache by the Tensor Data Mover (tensor_load_to_lds, 2D D#,
// TENSORcnt), issued once by wave 0.  Vertically-clipped halo rows are
// zeroed in LDS, so the K-loop im2col needs no y-bounds check
// (cache row index == kh).
//
// K loop is software-pipelined with double-buffered LDS fragment tiles:
// chunk k+32 is staged (A: LDS row-cache swizzle, B: global b128 weight
// loads) while WMMAs consume chunk k; single barrier per iteration.
//
// Fragments are stored pre-swizzled into WMMA wave32 lane order:
//   A frag: lane L -> row m = L%16, khalf = L/16;
//           element (v,e): k = khalf*16 + v*2 + e
//   B frag: lane L -> col n = L%16; lanes 0-15 k=0..15, lanes 16-31 k=16..31
// -------------------------------------------------------------------------
template<int KH, int KW, int CIN, bool HAS_BIAS>
__global__ __launch_bounds__(512) void conv_wmma_kernel(
    const unsigned short* __restrict__ in,    // bf16 [Nimg,64,64,CIN]
    const unsigned short* __restrict__ wtsT,  // bf16 [256, KLEN]
    const float* __restrict__ bias,           // [256] or nullptr
    float* __restrict__ out)                  // f32 [Nimg,64,64,256]
{
  constexpr int Hh = 64, Ww = 64, NG = 256;
  constexpr int KLEN = KH * KW * CIN;
  constexpr int PH = KH / 2, PW = KW / 2;
  constexpr int ROWE = Ww * CIN;             // bf16 elements per image row

  __shared__ __align__(16) unsigned short lIn[KH * ROWE];      // row cache
  __shared__ __align__(16) unsigned short lA[2][4][32][16];    // 8 KB
  __shared__ __align__(16) unsigned short lB[2][16][32][16];   // 32 KB

  const int tid  = threadIdx.x;
  const int lane = tid & 31;
  const int wv   = tid >> 5;    // 0..15
  const int wm   = wv >> 2;     // 0..3 : wave M tile
  const int wn   = wv & 3;      // 0..3 : wave N tile

  const long p0 = (long)blockIdx.x << 6;   // 64 pixels = one image row
  const int  img = (int)(p0 >> 12);
  const int  y   = (int)((p0 >> 6) & 63);
  const unsigned short* inImg = in + (long)img * (Hh * Ww * CIN);

  // ---------------- stage the KH input rows into LDS ----------------
  const int y0 = y - PH;
  const int topClip = (y0 < 0) ? -y0 : 0;
  const int botClip = (y0 + KH > Hh) ? (y0 + KH - Hh) : 0;
  const int nvalid  = KH - topClip - botClip;

  if (topClip) {
    unsigned* pz = (unsigned*)lIn;
    for (int i = tid; i < topClip * (ROWE / 2); i += 512) pz[i] = 0u;
  }
  if (botClip) {
    unsigned* pz = (unsigned*)(lIn + (size_t)(KH - botClip) * ROWE);
    for (int i = tid; i < botClip * (ROWE / 2); i += 512) pz[i] = 0u;
  }

  if (wv == 0) {
    // Tensor DMA: 2D tile (nvalid rows x ROWE bf16), contiguous rows.
    const unsigned long gaddr =
        (unsigned long)(const void*)(inImg + (long)(y0 + topClip) * ROWE);
    const unsigned ldsOff =
        (unsigned)(unsigned long)(const void*)(lIn + (size_t)topClip * ROWE);
    v4u g0; v8u g1;
    g0[0] = 1u;                                   // count=1, user mode
    g0[1] = ldsOff;                               // lds_addr (bytes)
    g0[2] = (unsigned)gaddr;                      // global_addr[31:0]
    g0[3] = (unsigned)((gaddr >> 32) & 0x01FFFFFFu) | (2u << 30); // hi | type=2
    g1[0] = 0x00010000u;                          // wg_mask=0, data_size=1 (2B)
    g1[1] = ((unsigned)ROWE & 0xFFFFu) << 16;     // tensor_dim0[15:0]
    g1[2] = (((unsigned)ROWE >> 16) & 0xFFFFu)    // tensor_dim0[31:16]
          | (((unsigned)nvalid & 0xFFFFu) << 16); // tensor_dim1[15:0]
    g1[3] = ((unsigned)ROWE & 0xFFFFu) << 16;     // tile_dim0
    g1[4] = (unsigned)nvalid & 0xFFFFu;           // tile_dim1 (tile_dim2=0)
    g1[5] = (unsigned)ROWE;                       // tensor_dim0_stride lo
    g1[6] = 0u;                                   // stride hi / dim1_stride lo
    g1[7] = 0u;
    asm volatile("tensor_load_to_lds %0, %1" :: "s"(g0), "s"(g1) : "memory");
    __builtin_amdgcn_s_wait_tensorcnt(0);
  }
  __syncthreads();

  // ---------------- staging thread roles ----------------
  // A: 4 frags x 32 lanes x 16 bf16 = 2048 elems; 4 per thread (one b64 store)
  const int af     = tid >> 7;                 // frag 0..3
  const int ar     = tid & 127;
  const int alane  = ar >> 2;                  // 0..31
  const int aq     = ar & 3;                   // quarter of the 16 elems
  const int am     = (af << 4) + (alane & 15); // pixel in block == x coord
  const int akhalf = alane >> 4;

  // B: 16 frags x 32 lanes x 16 bf16 = 8192 elems; 16 per thread (2 b128)
  const int bnf = tid >> 5;                    // 0..15
  const int bn  = (bnf << 4) + (lane & 15);    // output channel 0..255
  const int bkh = (lane >> 4) << 4;            // 0 or 16
  const uint4* bp = (const uint4*)(wtsT + (long)bn * KLEN + bkh);

  auto stageA = [&](int kk0, int sel) {
    uint2 av;
#pragma unroll
    for (int jj = 0; jj < 2; ++jj) {
      const int ee = (aq << 2) + (jj << 1);    // even element index
      const int v  = ee >> 1;
      const int k  = kk0 + (akhalf << 4) + ee;
      const int c    = k % CIN;                // even; c,c+1 same tap
      const int ridx = k / CIN;
      const int kh = ridx / KW;
      const int kw = ridx % KW;
      const int ix = am + kw - PW;
      unsigned val = 0u;
      if ((unsigned)ix < (unsigned)Ww)
        val = *(const unsigned*)&lIn[(size_t)(kh * Ww + ix) * CIN + c];
      ((unsigned*)&av)[jj] = val;
    }
    *(uint2*)(&lA[sel][af][alane][aq << 2]) = av;
  };
  auto stageB = [&](int kk0, int sel) {
    ((uint4*)(&lB[sel][bnf][lane][0]))[0] = bp[(kk0 >> 3) + 0];
    ((uint4*)(&lB[sel][bnf][lane][0]))[1] = bp[(kk0 >> 3) + 1];
  };

  v8f acc[4] = {};

  // ---------------- software-pipelined K loop ----------------
  stageA(0, 0);
  stageB(0, 0);
  __syncthreads();

  for (int kk0 = 0; kk0 < KLEN; kk0 += 32) {
    const int cur = (kk0 >> 5) & 1;
    if (kk0 + 32 < KLEN) {          // stage next chunk into the other buffer
      stageB(kk0 + 32, cur ^ 1);    // global weight loads issued early
      stageA(kk0 + 32, cur ^ 1);    // LDS row-cache im2col swizzle
    }

    const v16bf a = *(const v16bf*)(&lA[cur][wm][lane][0]);
#pragma unroll
    for (int nf = 0; nf < 4; ++nf) {
      const v16bf b = *(const v16bf*)(&lB[cur][(wn << 2) + nf][lane][0]);
      acc[nf] = __builtin_amdgcn_wmma_f32_16x16x32_bf16(
          false, a, false, b, (short)0, acc[nf], false, false);
    }
    __syncthreads();                // one barrier per K step
  }

  // ---- epilogue: C layout — VGPR i, lanes 0-15: m=i, n=lane; 16-31: m=8+i
  const int mh   = (lane >> 4) << 3;
  const int ncol = lane & 15;
#pragma unroll
  for (int nf = 0; nf < 4; ++nf) {
    const int n = (wn << 6) + (nf << 4) + ncol;
    const float bb = HAS_BIAS ? bias[n] : 0.0f;
#pragma unroll
    for (int i = 0; i < 8; ++i) {
      const long m = p0 + (wm << 4) + mh + i;
      out[m * NG + n] = acc[nf][i] + bb;
    }
  }
}

// -------------------------------------------------------------------------
// Fused LSTM cell: z = xz(b*T+t) + rz(b); gates i,f,g,o; keras hard-sigmoid.
// Writes fp32 cell state, bf16 h (next recurrent conv input), and relu'd
// sequence output (bf16 layer-1 chaining, or fp32 final output).
// -------------------------------------------------------------------------
template<bool OUT_F32>
__global__ __launch_bounds__(256) void lstm_cell_kernel(
    const float* __restrict__ xz,        // [B*T,64,64,256] (bias included)
    const float* __restrict__ rz,        // [B,64,64,256]
    float* __restrict__ c,               // [B,64,64,64]
    unsigned short* __restrict__ h,      // bf16 [B,64,64,64]
    void* __restrict__ seq,              // [B*T,64,64,64] bf16 or f32
    int t)
{
  const int idx = blockIdx.x * 256 + threadIdx.x;   // < 4*4096*64
  const int f   = idx & 63;
  const int pix = idx >> 6;
  const int b   = pix >> 12;
  const int yx  = pix & 4095;
  const long img = (long)b * 8 + t;                 // x layout: (b, t)
  const long zb  = ((img << 12) + yx) << 8;
  const long rb  = (long)pix << 8;

  const float zi = xz[zb +       f] + rz[rb +       f];
  const float zf = xz[zb +  64 + f] + rz[rb +  64 + f];
  const float zg = xz[zb + 128 + f] + rz[rb + 128 + f];
  const float zo = xz[zb + 192 + f] + rz[rb + 192 + f];

  const float cn = hsig(zf) * c[idx] + hsig(zi) * tanhf(zg);
  const float hn = hsig(zo) * tanhf(cn);
  c[idx] = cn;
  h[idx] = f32_to_bf16(hn);

  const float r = fmaxf(hn, 0.0f);
  const long so = (((img << 12) + yx) << 6) + f;
  if (OUT_F32) ((float*)seq)[so] = r;
  else         ((unsigned short*)seq)[so] = f32_to_bf16(r);
}

// -------------------------------------------------------------------------
// Host-side orchestration
// -------------------------------------------------------------------------
extern "C" void kernel_launch(void* const* d_in, const int* in_sizes, int n_in,
                              void* d_out, int out_size, void* d_ws, size_t ws_size,
                              hipStream_t stream) {
  (void)in_sizes; (void)n_in; (void)out_size; (void)ws_size;

  const float* x  = (const float*)d_in[0];
  const float* K1 = (const float*)d_in[1];
  const float* R1 = (const float*)d_in[2];
  const float* b1 = (const float*)d_in[3];
  const float* K2 = (const float*)d_in[4];
  const float* R2 = (const float*)d_in[5];
  const float* b2 = (const float*)d_in[6];
  float* out = (float*)d_out;

  constexpr int Bq = 4, T = 8, HW = 64 * 64, F = 64;
  constexpr int NSEQ = Bq * T;              // 32 images
  constexpr int KL_K1 = 5 * 5 * 32;         // 800
  constexpr int KL_R1 = 5 * 5 * 64;         // 1600
  constexpr int KL_2  = 3 * 3 * 64;         // 576

  char* ws = (char*)d_ws;
  size_t off = 0;
  auto alloc = [&](size_t bytes) -> void* {
    void* p = ws + off;
    off = (off + bytes + 255) & ~(size_t)255;
    return p;
  };

  unsigned short* x_bf = (unsigned short*)alloc((size_t)NSEQ * HW * 32 * 2);
  unsigned short* wK1  = (unsigned short*)alloc((size_t)KL_K1 * 256 * 2);
  unsigned short* wR1  = (unsigned short*)alloc((size_t)KL_R1 * 256 * 2);
  unsigned short* wK2  = (unsigned short*)alloc((size_t)KL_2 * 256 * 2);
  unsigned short* wR2  = (unsigned short*)alloc((size_t)KL_2 * 256 * 2);
  float*          xz   = (float*)alloc((size_t)NSEQ * HW * 256 * 4);   // 128 MB
  float*          rz   = (float*)alloc((size_t)Bq * HW * 256 * 4);     // 16 MB
  unsigned short* hbuf = (unsigned short*)alloc((size_t)Bq * HW * F * 2);
  float*          cbuf = (float*)alloc((size_t)Bq * HW * F * 4);
  unsigned short* y1   = (unsigned short*)alloc((size_t)NSEQ * HW * F * 2);

  // --- one-time conversions ---
  {
    int n = NSEQ * HW * 32;
    cast_f32_to_bf16_kernel<<<(n + 255) / 256, 256, 0, stream>>>(x, x_bf, n);
    transpose_w_to_bf16_kernel<<<(KL_K1 * 256 + 255) / 256, 256, 0, stream>>>(K1, wK1, KL_K1);
    transpose_w_to_bf16_kernel<<<(KL_R1 * 256 + 255) / 256, 256, 0, stream>>>(R1, wR1, KL_R1);
    transpose_w_to_bf16_kernel<<<(KL_2  * 256 + 255) / 256, 256, 0, stream>>>(K2, wK2, KL_2);
    transpose_w_to_bf16_kernel<<<(KL_2  * 256 + 255) / 256, 256, 0, stream>>>(R2, wR2, KL_2);
  }

  const int gridSeq = NSEQ * HW / 64;                // 2048 blocks
  const int gridRec = Bq * HW / 64;                  // 256 blocks
  const int cellBlocks = Bq * HW * F / 256;          // 4096
  const int hWords = Bq * HW * F / 2;                // bf16 h as u32 words
  const int cWords = Bq * HW * F;

  // ================= Layer 1 =================
  conv_wmma_kernel<5, 5, 32, true><<<gridSeq, 512, 0, stream>>>(x_bf, wK1, b1, xz);
  fill_zero_kernel<<<(hWords + 255) / 256, 256, 0, stream>>>((unsigned int*)hbuf, hWords);
  fill_zero_kernel<<<(cWords + 255) / 256, 256, 0, stream>>>((unsigned int*)cbuf, cWords);
  for (int t = 0; t < T; ++t) {
    conv_wmma_kernel<5, 5, 64, false><<<gridRec, 512, 0, stream>>>(hbuf, wR1, nullptr, rz);
    lstm_cell_kernel<false><<<cellBlocks, 256, 0, stream>>>(xz, rz, cbuf, hbuf, (void*)y1, t);
  }

  // ================= Layer 2 =================
  conv_wmma_kernel<3, 3, 64, true><<<gridSeq, 512, 0, stream>>>(y1, wK2, b2, xz);
  fill_zero_kernel<<<(hWords + 255) / 256, 256, 0, stream>>>((unsigned int*)hbuf, hWords);
  fill_zero_kernel<<<(cWords + 255) / 256, 256, 0, stream>>>((unsigned int*)cbuf, cWords);
  for (int t = 0; t < T; ++t) {
    conv_wmma_kernel<3, 3, 64, false><<<gridRec, 512, 0, stream>>>(hbuf, wR2, nullptr, rz);
    lstm_cell_kernel<true><<<cellBlocks, 256, 0, stream>>>(xz, rz, cbuf, hbuf, (void*)out, t);
  }
}
